// RNNDecoder_68109591380155
// MI455X (gfx1250) — matrix operand order repaired and checked by reading, not compiled, not run on hardware
//
#include <hip/hip_runtime.h>
#include <hip/hip_bf16.h>

// ---------------------------------------------------------------------------
// NRI-style recurrent graph decoder, fused for MI455X (gfx1250, wave32, WMMA)
// All GEMMs on v_wmma_f32_16x16x32_f16. One workgroup per (batch, receiver)
// for message passing -> scatter becomes an LDS reduction (edges are
// receiver-major: e = r*63 + j, send = j + (j>=r)).
// This revision: 2-way N-tile unroll -> two independent WMMA accumulator
// chains per inner loop (hides XDL latency + load waits).
// ---------------------------------------------------------------------------

typedef __attribute__((ext_vector_type(16))) _Float16 v16h;
typedef __attribute__((ext_vector_type(8)))  _Float16 v8h;
typedef __attribute__((ext_vector_type(8)))  float    v8f;

#define WMMA(a,b,c) __builtin_amdgcn_wmma_f32_16x16x32_f16(false,(a),false,(b),(short)0,(c),false,false)

namespace {
constexpr int B_  = 16;
constexpr int N_  = 64;
constexpr int ET_ = 3;
constexpr int H_  = 256;
constexpr int D_  = 4;
constexpr int NC_ = 3;
constexpr int S_  = 8;
constexpr int E_  = N_ * (N_ - 1);   // 4032
constexpr int H2_ = 2 * H_;          // 512
constexpr int ST  = 264;             // padded LDS row stride (halves), 528B = 33*16B
}

// A-fragment (16x32, f16): lane<16 holds K {k0..k0+7, k0+16..k0+23},
// lane>=16 holds K {k0+8..15, k0+24..31}; M = lane%16 selected by rowbase.
__device__ __forceinline__ v16h afrag(const _Float16* rowbase, int laneHi) {
  const _Float16* p = rowbase + laneHi * 8;
  v8h x = *(const v8h*)p;
  v8h y = *(const v8h*)(p + 16);
  v16h r;
#pragma unroll
  for (int i = 0; i < 8; ++i) { r[i] = x[i]; r[i + 8] = y[i]; }
  return r;
}

// B-fragment (32x16, f16) for D = A * W^T with W row-major [n, k]:
// lane n = lane%16 holds W row (n0+n); lane<16 K {k0..k0+7, k0+16..k0+23}, lane>=16 K {k0+8..15, k0+24..31}.
__device__ __forceinline__ v16h bfrag(const _Float16* w, int ldk, int n0, int k0,
                                      int lane16, int laneHi) {
  const _Float16* p = w + (size_t)(n0 + lane16) * ldk + k0 + laneHi * 8;
  v8h x = *(const v8h*)p;
  v8h y = *(const v8h*)(p + 16);
  v16h r;
#pragma unroll
  for (int i = 0; i < 8; ++i) { r[i] = x[i]; r[i + 8] = y[i]; }
  return r;
}

__device__ __forceinline__ float sigmoidf_(float x) { return 1.0f / (1.0f + __expf(-x)); }

// ---------------------------------------------------------------------------
// fp32 -> fp16 weight conversion
// ---------------------------------------------------------------------------
__global__ void cvt_kernel(const float* __restrict__ src, _Float16* __restrict__ dst, int n) {
  int i = blockIdx.x * 256 + threadIdx.x;
  if (i < n) dst[i] = (_Float16)src[i];
}

// ---------------------------------------------------------------------------
// Message passing + aggregation. grid = B*N blocks, 128 threads (4 waves).
// Block (b, r): 63 incoming edges (padded to 64). Wave w owns edges
// [16w, 16w+16). Layer1: pre[64,512] @ w1_t[256,512]^T, tanh.
// Layer2: m1[64,256] @ w2_t[256,256]^T, tanh, * rel_type/(ET*N), LDS-reduce
// over edges -> agg16[b*N+r, 256].
// ---------------------------------------------------------------------------
__global__ __launch_bounds__(128)
void msg_kernel(const _Float16* __restrict__ h16,
                const _Float16* __restrict__ w1,   // [ET,256,512] f16
                const float*    __restrict__ b1,   // [ET,256]
                const _Float16* __restrict__ w2,   // [ET,256,256] f16
                const float*    __restrict__ b2,   // [ET,256]
                const float*    __restrict__ rt,   // [B,E,ET]
                _Float16*       __restrict__ agg16) // [B*N,256]
{
  __shared__ _Float16 sRecv[256];
  __shared__ _Float16 sSend[64 * ST];
  __shared__ _Float16 sM1[4 * 16 * ST];
  __shared__ float    sRt[ET_ * 64];
  __shared__ float    sAgg[H_];

  const int tid    = threadIdx.x;
  const int wave   = tid >> 5;
  const int lane   = tid & 31;
  const int lane16 = lane & 15;
  const int laneHi = lane >> 4;
  const int b      = blockIdx.x >> 6;
  const int r      = blockIdx.x & 63;

  // receiver row (shared by all 63 edges of this block)
  for (int i = tid; i < H_ / 8; i += 128)
    ((v8h*)sRecv)[i] = ((const v8h*)(h16 + (size_t)(b * N_ + r) * H_))[i];

  // sender rows (row 63 = zero padding, killed later by rt weight 0)
  for (int i = tid; i < 64 * (H_ / 8); i += 128) {
    int j = i >> 5;          // edge slot 0..63
    int c = i & 31;          // 8-half chunk
    v8h v = {};
    if (j < 63) {
      int snd = j + (j >= r ? 1 : 0);
      v = ((const v8h*)(h16 + (size_t)(b * N_ + snd) * H_))[c];
    }
    *(v8h*)(sSend + j * ST + c * 8) = v;
  }

  // rel_type weights with 1/(ET*N) folded in
  for (int i = tid; i < ET_ * 64; i += 128) {
    int t = i >> 6, j = i & 63;
    sRt[t * 64 + j] = (j < 63)
        ? rt[((size_t)b * E_ + (size_t)r * 63 + j) * ET_ + t] * (1.0f / (ET_ * N_))
        : 0.0f;
  }
  for (int i = tid; i < H_; i += 128) sAgg[i] = 0.0f;
  __syncthreads();

  const int m0 = wave * 16;            // edge-tile base for this wave
  _Float16* myM1 = sM1 + wave * 16 * ST;

  for (int t = 0; t < ET_; ++t) {
    const _Float16* w1t = w1 + (size_t)t * H_ * H2_;
    const _Float16* w2t = w2 + (size_t)t * H_ * H_;
    const float*    b1t = b1 + t * H_;
    const float*    b2t = b2 + t * H_;

    // ---- layer 1: [16,512] x [256,512]^T, two N-tiles per pass ----
    for (int nt = 0; nt < 16; nt += 2) {
      const int n0 = nt * 16;
      const int n1 = n0 + 16;
      v8f acc0 = {}, acc1 = {};
#pragma unroll
      for (int kb = 0; kb < 8; ++kb) {           // K 0..255: recv half (same row all edges)
        v16h a   = afrag(sRecv + kb * 32, laneHi);
        v16h bf0 = bfrag(w1t, H2_, n0, kb * 32, lane16, laneHi);
        v16h bf1 = bfrag(w1t, H2_, n1, kb * 32, lane16, laneHi);
        acc0 = WMMA(a, bf0, acc0);
        acc1 = WMMA(a, bf1, acc1);
      }
      const _Float16* arow = sSend + (m0 + lane16) * ST;
#pragma unroll
      for (int kb = 0; kb < 8; ++kb) {           // K 256..511: send half
        v16h a   = afrag(arow + kb * 32, laneHi);
        v16h bf0 = bfrag(w1t, H2_, n0, 256 + kb * 32, lane16, laneHi);
        v16h bf1 = bfrag(w1t, H2_, n1, 256 + kb * 32, lane16, laneHi);
        acc0 = WMMA(a, bf0, acc0);
        acc1 = WMMA(a, bf1, acc1);
      }
      float bias0 = b1t[n0 + lane16];
      float bias1 = b1t[n1 + lane16];
#pragma unroll
      for (int v = 0; v < 8; ++v) {
        int m = v + 8 * laneHi;
        myM1[m * ST + n0 + lane16] = (_Float16)tanhf(acc0[v] + bias0);
        myM1[m * ST + n1 + lane16] = (_Float16)tanhf(acc1[v] + bias1);
      }
    }

    // ---- layer 2: [16,256] x [256,256]^T, two N-tiles per pass ----
    for (int nt = 0; nt < 16; nt += 2) {
      const int n0 = nt * 16;
      const int n1 = n0 + 16;
      v8f acc0 = {}, acc1 = {};
      const _Float16* arow = myM1 + lane16 * ST;
#pragma unroll
      for (int kb = 0; kb < 8; ++kb) {
        v16h a   = afrag(arow + kb * 32, laneHi);
        v16h bf0 = bfrag(w2t, H_, n0, kb * 32, lane16, laneHi);
        v16h bf1 = bfrag(w2t, H_, n1, kb * 32, lane16, laneHi);
        acc0 = WMMA(a, bf0, acc0);
        acc1 = WMMA(a, bf1, acc1);
      }
      float bias0 = b2t[n0 + lane16];
      float bias1 = b2t[n1 + lane16];
      float partial0 = 0.0f, partial1 = 0.0f;
#pragma unroll
      for (int v = 0; v < 8; ++v) {
        int m = v + 8 * laneHi;
        float w = sRt[t * 64 + m0 + m];
        partial0 += tanhf(acc0[v] + bias0) * w;
        partial1 += tanhf(acc1[v] + bias1) * w;
      }
      atomicAdd(&sAgg[n0 + lane16], partial0);
      atomicAdd(&sAgg[n1 + lane16], partial1);
    }
  }
  __syncthreads();

  for (int i = tid; i < H_; i += 128)
    agg16[(size_t)blockIdx.x * H_ + i] = (_Float16)sAgg[i];
}

// ---------------------------------------------------------------------------
// GRU update. grid = 16 blocks x 128 threads; wave owns a 16-row tile of
// [B*N=1024, H]. Three gate GEMMs [16,256]x[256,256]^T (3 independent
// accumulator chains) + tiny D=4 input lins.
// ---------------------------------------------------------------------------
__global__ __launch_bounds__(128)
void gru_kernel(const _Float16* __restrict__ agg16,   // [1024,256]
                const _Float16* __restrict__ hidw16,  // [3,256,256]: r,i,h
                const float* __restrict__ wr, const float* __restrict__ br,
                const float* __restrict__ wi, const float* __restrict__ bi,
                const float* __restrict__ wn, const float* __restrict__ bn,
                const float* __restrict__ data,       // [B,N,S,D]
                const float* __restrict__ prev,       // [B*N,D]
                const int*   __restrict__ burn_in,
                int s,
                float*    __restrict__ h,             // [1024,256]
                _Float16* __restrict__ h16)
{
  __shared__ float sIns[64][D_];

  const int tid    = threadIdx.x;
  const int wave   = tid >> 5;
  const int lane   = tid & 31;
  const int lane16 = lane & 15;
  const int laneHi = lane >> 4;
  const int row0   = blockIdx.x * 64;
  const bool useData = (s < burn_in[0]);

  for (int i = tid; i < 64 * D_; i += 128) {
    int rr = i >> 2, d = i & 3;
    int grow = row0 + rr;
    int bb = grow >> 6, nn = grow & 63;
    sIns[rr][d] = useData ? data[(((size_t)bb * N_ + nn) * S_ + s) * D_ + d]
                          : prev[(size_t)grow * D_ + d];
  }
  __syncthreads();

  const int m0 = row0 + wave * 16;
  v16h af[8];
  const _Float16* arow = agg16 + (size_t)(m0 + lane16) * H_;
#pragma unroll
  for (int kb = 0; kb < 8; ++kb) af[kb] = afrag(arow + kb * 32, laneHi);

  for (int nt = 0; nt < 16; ++nt) {
    const int n0 = nt * 16;
    v8f ar = {}, ai = {}, ah = {};
#pragma unroll
    for (int kb = 0; kb < 8; ++kb) {
      v16h b0 = bfrag(hidw16 + 0 * H_ * H_, H_, n0, kb * 32, lane16, laneHi);
      v16h b1 = bfrag(hidw16 + 1 * H_ * H_, H_, n0, kb * 32, lane16, laneHi);
      v16h b2 = bfrag(hidw16 + 2 * H_ * H_, H_, n0, kb * 32, lane16, laneHi);
      ar = WMMA(af[kb], b0, ar);
      ai = WMMA(af[kb], b1, ai);
      ah = WMMA(af[kb], b2, ah);
    }
    const int hc = n0 + lane16;
    const float wr0 = wr[hc*4+0], wr1 = wr[hc*4+1], wr2 = wr[hc*4+2], wr3 = wr[hc*4+3];
    const float wi0 = wi[hc*4+0], wi1 = wi[hc*4+1], wi2 = wi[hc*4+2], wi3 = wi[hc*4+3];
    const float wn0 = wn[hc*4+0], wn1 = wn[hc*4+1], wn2 = wn[hc*4+2], wn3 = wn[hc*4+3];
    const float bR = br[hc], bI = bi[hc], bN = bn[hc];
#pragma unroll
    for (int v = 0; v < 8; ++v) {
      int m = v + 8 * laneHi;
      int lrow = wave * 16 + m;
      int grow = m0 + m;
      float i0 = sIns[lrow][0], i1 = sIns[lrow][1], i2 = sIns[lrow][2], i3 = sIns[lrow][3];
      float linr = bR + i0*wr0 + i1*wr1 + i2*wr2 + i3*wr3;
      float lini = bI + i0*wi0 + i1*wi1 + i2*wi2 + i3*wi3;
      float linn = bN + i0*wn0 + i1*wn1 + i2*wn2 + i3*wn3;
      float rg = sigmoidf_(linr + ar[v]);
      float ig = sigmoidf_(lini + ai[v]);
      float ng = tanhf(linn + rg * ah[v]);
      size_t off = (size_t)grow * H_ + hc;
      float hn = (1.0f - ig) * ng + ig * h[off];
      h[off]   = hn;
      h16[off] = (_Float16)hn;
    }
  }
}

// ---------------------------------------------------------------------------
// Output head. grid = 16 blocks x 128 threads; wave owns a 16-row tile.
// pred1 = relu(cat(hg,hhg) @ out1^T), pred2 = relu(pred1 @ out2^T),
// then alpha/mu heads + softmax/argmax + outputs + next-step input.
// Two N-tiles per pass for WMMA ILP.
// ---------------------------------------------------------------------------
__global__ __launch_bounds__(128)
void out_kernel(const _Float16* __restrict__ h16g, const _Float16* __restrict__ h16h,
                const _Float16* __restrict__ o1w, const float* __restrict__ o1b,
                const _Float16* __restrict__ o2w, const float* __restrict__ o2b,
                const float* __restrict__ aw, const float* __restrict__ ab,
                const float* __restrict__ mw, const float* __restrict__ mb,
                const float* __restrict__ data,
                float* __restrict__ prev,
                const int* __restrict__ burn_in,
                int s,
                float* __restrict__ out)
{
  __shared__ _Float16 sP1[4 * 16 * ST];
  __shared__ _Float16 sP2[4 * 16 * ST];

  const int tid    = threadIdx.x;
  const int wave   = tid >> 5;
  const int lane   = tid & 31;
  const int lane16 = lane & 15;
  const int laneHi = lane >> 4;
  const int m0     = blockIdx.x * 64 + wave * 16;

  _Float16* p1 = sP1 + wave * 16 * ST;
  _Float16* p2 = sP2 + wave * 16 * ST;

  // stage 1: [16,512] x [256,512]^T, relu
  for (int nt = 0; nt < 16; nt += 2) {
    const int n0 = nt * 16;
    const int n1 = n0 + 16;
    v8f acc0 = {}, acc1 = {};
    const _Float16* ag = h16g + (size_t)(m0 + lane16) * H_;
    const _Float16* ah = h16h + (size_t)(m0 + lane16) * H_;
#pragma unroll
    for (int kb = 0; kb < 8; ++kb) {
      v16h a   = afrag(ag + kb * 32, laneHi);
      v16h bf0 = bfrag(o1w, H2_, n0, kb * 32, lane16, laneHi);
      v16h bf1 = bfrag(o1w, H2_, n1, kb * 32, lane16, laneHi);
      acc0 = WMMA(a, bf0, acc0);
      acc1 = WMMA(a, bf1, acc1);
    }
#pragma unroll
    for (int kb = 0; kb < 8; ++kb) {
      v16h a   = afrag(ah + kb * 32, laneHi);
      v16h bf0 = bfrag(o1w, H2_, n0, 256 + kb * 32, lane16, laneHi);
      v16h bf1 = bfrag(o1w, H2_, n1, 256 + kb * 32, lane16, laneHi);
      acc0 = WMMA(a, bf0, acc0);
      acc1 = WMMA(a, bf1, acc1);
    }
    float bias0 = o1b[n0 + lane16];
    float bias1 = o1b[n1 + lane16];
#pragma unroll
    for (int v = 0; v < 8; ++v) {
      int m = v + 8 * laneHi;
      p1[m * ST + n0 + lane16] = (_Float16)fmaxf(acc0[v] + bias0, 0.0f);
      p1[m * ST + n1 + lane16] = (_Float16)fmaxf(acc1[v] + bias1, 0.0f);
    }
  }

  // stage 2: [16,256] x [256,256]^T, relu
  for (int nt = 0; nt < 16; nt += 2) {
    const int n0 = nt * 16;
    const int n1 = n0 + 16;
    v8f acc0 = {}, acc1 = {};
    const _Float16* arow = p1 + lane16 * ST;
#pragma unroll
    for (int kb = 0; kb < 8; ++kb) {
      v16h a   = afrag(arow + kb * 32, laneHi);
      v16h bf0 = bfrag(o2w, H_, n0, kb * 32, lane16, laneHi);
      v16h bf1 = bfrag(o2w, H_, n1, kb * 32, lane16, laneHi);
      acc0 = WMMA(a, bf0, acc0);
      acc1 = WMMA(a, bf1, acc1);
    }
    float bias0 = o2b[n0 + lane16];
    float bias1 = o2b[n1 + lane16];
#pragma unroll
    for (int v = 0; v < 8; ++v) {
      int m = v + 8 * laneHi;
      p2[m * ST + n0 + lane16] = (_Float16)fmaxf(acc0[v] + bias0, 0.0f);
      p2[m * ST + n1 + lane16] = (_Float16)fmaxf(acc1[v] + bias1, 0.0f);
    }
  }

  // stage 3: per-row heads (lanes 0..15, one row each)
  if (lane < 16) {
    const int grow = m0 + lane;
    const _Float16* pr = p2 + lane * ST;

    float alpha[NC_], mu[NC_ * D_];
#pragma unroll
    for (int c = 0; c < NC_; ++c) {
      float sdot = ab[c];
      for (int k = 0; k < H_; ++k) sdot += (float)pr[k] * aw[c * H_ + k];
      alpha[c] = sdot;
    }
#pragma unroll
    for (int c = 0; c < NC_ * D_; ++c) {
      float sdot = mb[c];
      for (int k = 0; k < H_; ++k) sdot += (float)pr[k] * mw[c * H_ + k];
      mu[c] = sdot;
    }
    float mx = fmaxf(alpha[0], fmaxf(alpha[1], alpha[2]));
    float e0 = __expf(alpha[0] - mx), e1 = __expf(alpha[1] - mx), e2 = __expf(alpha[2] - mx);
    float inv = 1.0f / (e0 + e1 + e2);
    float a0 = e0 * inv, a1 = e1 * inv, a2 = e2 * inv;
    int idx = 0; float best = a0;
    if (a1 > best) { best = a1; idx = 1; }
    if (a2 > best) { best = a2; idx = 2; }

    const int bb = grow >> 6, nn = grow & 63;
    const bool useData = (s < burn_in[0]);
    float ins[D_];
#pragma unroll
    for (int d = 0; d < D_; ++d)
      ins[d] = useData ? data[(((size_t)bb * N_ + nn) * S_ + s) * D_ + d]
                       : prev[(size_t)grow * D_ + d];
    float nxt[D_];
#pragma unroll
    for (int d = 0; d < D_; ++d) nxt[d] = ins[d] + mu[idx * D_ + d];

    float* preds  = out;                        // [B,N,S,D]
    float* alphas = out + B_*N_*S_*D_;          // [B,N,S,NC]
    float* mus    = alphas + B_*N_*S_*NC_;      // [B,N,S,NC,D]
    float* sig    = mus + B_*N_*S_*NC_*D_;      // [B,N,S,NC,D]

    size_t pb = ((size_t)grow * S_ + s) * D_;
#pragma unroll
    for (int d = 0; d < D_; ++d) preds[pb + d] = nxt[d];
    size_t abse = ((size_t)grow * S_ + s) * NC_;
    alphas[abse + 0] = a0; alphas[abse + 1] = a1; alphas[abse + 2] = a2;
    size_t mbse = ((size_t)grow * S_ + s) * NC_ * D_;
#pragma unroll
    for (int c = 0; c < NC_ * D_; ++c) { mus[mbse + c] = mu[c] + ins[c & 3]; sig[mbse + c] = 1.0f; }
#pragma unroll
    for (int d = 0; d < D_; ++d) prev[(size_t)grow * D_ + d] = nxt[d];
  }
}

// ---------------------------------------------------------------------------
// Host side
// ---------------------------------------------------------------------------
enum {
  IN_DATA = 0, IN_RTG, IN_RRG, IN_RSG, IN_RTH, IN_RRH, IN_RSH,
  P_MSG1G_W, P_MSG1G_B, P_MSG2G_W, P_MSG2G_B,
  P_HIDR_G, P_HIDI_G, P_HIDH_G,
  P_INR_G_W, P_INR_G_B, P_INI_G_W, P_INI_G_B, P_INN_G_W, P_INN_G_B,
  P_MSG1H_W, P_MSG1H_B, P_MSG2H_W, P_MSG2H_B,
  P_HIDR_H, P_HIDI_H, P_HIDH_H,
  P_INR_H_W, P_INR_H_B, P_INI_H_W, P_INI_H_B, P_INN_H_W, P_INN_H_B,
  P_OUT1_W, P_OUT1_B, P_OUT2_W, P_OUT2_B,
  P_ALPHA_W, P_ALPHA_B, P_MU_W, P_MU_B,
  IN_OUTSTEPS, IN_BURNIN, IN_VAR
};

extern "C" void kernel_launch(void* const* d_in, const int* in_sizes, int n_in,
                              void* d_out, int out_size, void* d_ws, size_t ws_size,
                              hipStream_t stream) {
  (void)in_sizes; (void)n_in; (void)out_size; (void)ws_size;
  char* ws = (char*)d_ws;

  float*    h_g    = (float*)(ws + 0);              // 1,048,576 B
  float*    h_hg   = (float*)(ws + 1048576);        // 1,048,576 B
  _Float16* h16_g  = (_Float16*)(ws + 2097152);     //   524,288 B
  _Float16* h16_hg = (_Float16*)(ws + 2621440);     //   524,288 B
  _Float16* agg16  = (_Float16*)(ws + 3145728);     //   524,288 B
  float*    prev   = (float*)(ws + 3670016);        //    16,384 B
  const size_t wo  = 3686400;
  _Float16* w_msg1g = (_Float16*)(ws + wo + 0);
  _Float16* w_msg2g = (_Float16*)(ws + wo + 786432);
  _Float16* w_hidg  = (_Float16*)(ws + wo + 1179648);
  _Float16* w_msg1h = (_Float16*)(ws + wo + 1572864);
  _Float16* w_msg2h = (_Float16*)(ws + wo + 2359296);
  _Float16* w_hidh  = (_Float16*)(ws + wo + 2752512);
  _Float16* w_out1  = (_Float16*)(ws + wo + 3145728);
  _Float16* w_out2  = (_Float16*)(ws + wo + 3407872);

  // zero hidden states / prev / agg (deterministic each call)
  hipMemsetAsync(d_ws, 0, 3686400, stream);

  auto cvt = [&](int idx, _Float16* dst, int n) {
    cvt_kernel<<<(n + 255) / 256, 256, 0, stream>>>((const float*)d_in[idx], dst, n);
  };
  cvt(P_MSG1G_W, w_msg1g, ET_ * H_ * H2_);
  cvt(P_MSG2G_W, w_msg2g, ET_ * H_ * H_);
  cvt(P_HIDR_G,  w_hidg + 0 * H_ * H_, H_ * H_);
  cvt(P_HIDI_G,  w_hidg + 1 * H_ * H_, H_ * H_);
  cvt(P_HIDH_G,  w_hidg + 2 * H_ * H_, H_ * H_);
  cvt(P_MSG1H_W, w_msg1h, ET_ * H_ * H2_);
  cvt(P_MSG2H_W, w_msg2h, ET_ * H_ * H_);
  cvt(P_HIDR_H,  w_hidh + 0 * H_ * H_, H_ * H_);
  cvt(P_HIDI_H,  w_hidh + 1 * H_ * H_, H_ * H_);
  cvt(P_HIDH_H,  w_hidh + 2 * H_ * H_, H_ * H_);
  cvt(P_OUT1_W,  w_out1, H_ * H2_);
  cvt(P_OUT2_W,  w_out2, H_ * H_);

  const float* data = (const float*)d_in[IN_DATA];
  const int* burnin = (const int*)d_in[IN_BURNIN];

  for (int s = 0; s < S_; ++s) {
    // graph "g"
    msg_kernel<<<B_ * N_, 128, 0, stream>>>(
        h16_g, w_msg1g, (const float*)d_in[P_MSG1G_B],
        w_msg2g, (const float*)d_in[P_MSG2G_B],
        (const float*)d_in[IN_RTG], agg16);
    gru_kernel<<<16, 128, 0, stream>>>(
        agg16, w_hidg,
        (const float*)d_in[P_INR_G_W], (const float*)d_in[P_INR_G_B],
        (const float*)d_in[P_INI_G_W], (const float*)d_in[P_INI_G_B],
        (const float*)d_in[P_INN_G_W], (const float*)d_in[P_INN_G_B],
        data, prev, burnin, s, h_g, h16_g);
    // graph "hg"
    msg_kernel<<<B_ * N_, 128, 0, stream>>>(
        h16_hg, w_msg1h, (const float*)d_in[P_MSG1H_B],
        w_msg2h, (const float*)d_in[P_MSG2H_B],
        (const float*)d_in[IN_RTH], agg16);
    gru_kernel<<<16, 128, 0, stream>>>(
        agg16, w_hidh,
        (const float*)d_in[P_INR_H_W], (const float*)d_in[P_INR_H_B],
        (const float*)d_in[P_INI_H_W], (const float*)d_in[P_INI_H_B],
        (const float*)d_in[P_INN_H_W], (const float*)d_in[P_INN_H_B],
        data, prev, burnin, s, h_hg, h16_hg);
    // output head (also writes prev for step s+1)
    out_kernel<<<16, 128, 0, stream>>>(
        h16_g, h16_hg, w_out1, (const float*)d_in[P_OUT1_B],
        w_out2, (const float*)d_in[P_OUT2_B],
        (const float*)d_in[P_ALPHA_W], (const float*)d_in[P_ALPHA_B],
        (const float*)d_in[P_MU_W], (const float*)d_in[P_MU_B],
        data, prev, burnin, s, (float*)d_out);
  }
}
